// ResonanceAttractor_88304527606101
// MI455X (gfx1250) — compile-verified, matrix operand order checked by reading, not run on hardware
//
#include <hip/hip_runtime.h>
#include <hip/hip_bf16.h>
#include <math.h>

// Problem constants (reference: signals [128, 4096] f32)
#define NB 128
#define NN 4096
#define TCH 128                 // shifts per t-chunk (per workgroup in grid.y)
#define NCHUNK (NN / TCH)       // 32
#define NROT 8                  // pre-rotated bf16 copies (alignment trick)
#define SD_ROWLEN (2 * NN)      // doubled rows for wrap-free reads
#define TG 4                    // shifts processed concurrently per wave

typedef __bf16 v16bf __attribute__((ext_vector_type(16)));
typedef float  v8f   __attribute__((ext_vector_type(8)));
typedef float  v2f   __attribute__((ext_vector_type(2)));

// ---------------------------------------------------------------------------
// Stage 0: f32 -> bf16, plus NROT pre-rotated doubled copies:
//   sd[c][j][n] = s_j[(n - c) mod N],  n in [0, 2N)
// With c = t mod 8 every WMMA B-fragment read is a contiguous, 16B-aligned
// global_load_b128 for any shift t.
// ---------------------------------------------------------------------------
__global__ void k_convert(const float* __restrict__ s,
                          __bf16* __restrict__ a,
                          __bf16* __restrict__ sd) {
  int idx = blockIdx.x * blockDim.x + threadIdx.x;
  if (idx >= NB * NN) return;
  int i = idx / NN, n = idx - i * NN;
  __bf16 bv = (__bf16)s[idx];
  a[idx] = bv;
#pragma unroll
  for (int c = 0; c < NROT; ++c) {
    __bf16* row = sd + ((size_t)c * NB + i) * SD_ROWLEN;
    int p = n + c;
    row[p] = bv;                              // covers [c, NN+c)
    if (p + NN < SD_ROWLEN) row[p + NN] = bv; // covers [NN+c, 2NN)
    if (p >= NN)            row[p - NN] = bv; // covers [0, c)
  }
}

// ISA 7.12.2 16-bit fragment: lane<16 holds K in {0..7,16..23}, lane>=16 holds
// K in {8..15,24..31} -> two contiguous 16B runs per fragment.
__device__ __forceinline__ v16bf load_frag(const __bf16* p) {
  v16bf f;
  float4 x0 = *(const float4*)(p);
  float4 x1 = *(const float4*)(p + 16);
  __builtin_memcpy(&f, &x0, 16);
  __builtin_memcpy((char*)&f + 16, &x1, 16);
  return f;
}

// ---------------------------------------------------------------------------
// Stage 1: cc[i,j,t] = sum_m s_i[m] * s_j[(m-t) mod N] via bf16 WMMA with a
// fused running (max, argmax-t). Each wave owns one 16x16 (i,j) tile and
// processes TG=4 shifts at once: one A fragment feeds 4 independent WMMA
// chains. Two-stage double buffering hides load latency with no
// register-rotation copies. The B stream is addressed as sd + integer offset
// (GEPs rooted at the kernel argument) so address-space inference keeps the
// loads as global_load_b128 (LOADcnt only), decoupled from the LDS A stream
// (DScnt) — loop-carried generic-pointer phis previously degraded these to
// flat_load_b128.
// grid = (8 i-tiles, 32 t-chunks), block = 256 = 8 waves (wave w = j-tile w).
// A block (16 x 4096 bf16 = 128 KB) staged in dynamic LDS (CDNA5: 320KB/WGP).
// ---------------------------------------------------------------------------
__global__ void __launch_bounds__(256)
k_ccmax(const __bf16* __restrict__ a,
        const __bf16* __restrict__ sd,
        float* __restrict__ pval,
        int*   __restrict__ pidx) {
  extern __shared__ __bf16 lds[];            // 16 rows x NN bf16
  const int ti   = blockIdx.x;               // i-tile 0..7
  const int wave = threadIdx.x >> 5;         // j-tile 0..7
  const int lane = threadIdx.x & 31;
  const int hl   = lane >> 4;                // lane half (ISA 16-bit layout)
  const int r16  = lane & 15;

  // Cooperative stage of the A row-block (contiguous region) into LDS.
  {
    const float4* src = (const float4*)(a + (size_t)ti * 16 * NN);
    float4*       dst = (float4*)lds;
    for (int x = threadIdx.x; x < (16 * NN) / 8; x += 256) dst[x] = src[x];
  }
  __syncthreads();

  const __bf16* aptr0 = lds + r16 * NN + 8 * hl;

  float bestv[8];
  int   bestt[8];
#pragma unroll
  for (int r = 0; r < 8; ++r) { bestv[r] = -3.4e38f; bestt[r] = 0; }

  const int jrow = wave * 16 + r16;
  const size_t sdRowBase = (size_t)jrow * SD_ROWLEN;

  for (int tg = 0; tg < TCH; tg += TG) {
    const int t0 = blockIdx.y * TCH + tg;

    // Per-shift B offsets from sd (copy c = t mod 8 keeps reads 16B-aligned).
    size_t boff[TG];
#pragma unroll
    for (int u = 0; u < TG; ++u) {
      const int t  = t0 + u;
      const int c  = t & (NROT - 1);
      const int tb = t - c;                  // multiple of 8
      boff[u] = ((size_t)c * NB) * SD_ROWLEN + sdRowBase
                + (size_t)(NN - tb) + (size_t)(8 * hl);
    }
    const __bf16* ap = aptr0;

    v8f acc[TG];
#pragma unroll
    for (int u = 0; u < TG; ++u) acc[u] = (v8f){};

    // Prologue: fragment set A <- k-step 0.
    v16bf aA = load_frag(ap);
    v16bf bA[TG];
#pragma unroll
    for (int u = 0; u < TG; ++u) bA[u] = load_frag(sd + boff[u]);
    v16bf aB;
    v16bf bB[TG];

    // Main loop over k-step pairs (2m, 2m+1), m = 0..62. Each half-iteration
    // loads the next step's fragments into the idle buffer set, then issues
    // the 4 independent WMMAs on the current set. Buffer roles swap twice per
    // iteration -> no loop-carried register copies.
#pragma unroll 1
    for (int m = 0; m < 63; ++m) {
      // Stage 1: load step 2m+1 into set B; WMMA on set A (step 2m).
      aB = load_frag(ap + 32);
#pragma unroll
      for (int u = 0; u < TG; ++u) bB[u] = load_frag(sd + boff[u] + 32);
#pragma unroll
      for (int u = 0; u < TG; ++u) {
        acc[u] = __builtin_amdgcn_wmma_f32_16x16x32_bf16(
            false, aA, false, bA[u], (short)0, acc[u], false, false);
      }
      // Stage 2: load step 2m+2 into set A; WMMA on set B (step 2m+1).
      aA = load_frag(ap + 64);
#pragma unroll
      for (int u = 0; u < TG; ++u) bA[u] = load_frag(sd + boff[u] + 64);
#pragma unroll
      for (int u = 0; u < TG; ++u) {
        acc[u] = __builtin_amdgcn_wmma_f32_16x16x32_bf16(
            false, aB, false, bB[u], (short)0, acc[u], false, false);
      }
      ap += 64;
#pragma unroll
      for (int u = 0; u < TG; ++u) boff[u] += 64;
    }
    // Epilogue: set A holds step 126; load step 127 into set B, finish both.
    aB = load_frag(ap + 32);
#pragma unroll
    for (int u = 0; u < TG; ++u) bB[u] = load_frag(sd + boff[u] + 32);
#pragma unroll
    for (int u = 0; u < TG; ++u) {
      acc[u] = __builtin_amdgcn_wmma_f32_16x16x32_bf16(
          false, aA, false, bA[u], (short)0, acc[u], false, false);
    }
#pragma unroll
    for (int u = 0; u < TG; ++u) {
      acc[u] = __builtin_amdgcn_wmma_f32_16x16x32_bf16(
          false, aB, false, bB[u], (short)0, acc[u], false, false);
    }

    // Running argmax in increasing-t order with strict '>' keeps the first
    // maximum, matching jnp.argmax semantics.
#pragma unroll
    for (int u = 0; u < TG; ++u) {
#pragma unroll
      for (int r = 0; r < 8; ++r) {
        float v = acc[u][r];
        if (v > bestv[r]) { bestv[r] = v; bestt[r] = t0 + u; }
      }
    }
  }

  // C/D layout: lane l, VGPR r -> M = r + 8*(l>>4), N = l&15.
#pragma unroll
  for (int r = 0; r < 8; ++r) {
    int i = ti * 16 + 8 * hl + r;
    int j = wave * 16 + r16;
    size_t o = ((size_t)blockIdx.y * NB + i) * NB + j;
    pval[o] = bestv[r];
    pidx[o] = bestt[r];
  }
}

// ---------------------------------------------------------------------------
// Stage 2: reduce (max, idx) over t-chunks; fm = |cc_peak| * sin(2*pi*peak/N);
// keep only upper triangle i<j.
// ---------------------------------------------------------------------------
__global__ void k_reduce(const float* __restrict__ pval,
                         const int*   __restrict__ pidx,
                         float* __restrict__ mup) {
  int idx = blockIdx.x * blockDim.x + threadIdx.x;
  if (idx >= NB * NB) return;
  int i = idx / NB, j = idx - i * NB;
  float best = -3.4e38f;
  int   bp   = 0;
  for (int ch = 0; ch < NCHUNK; ++ch) {       // increasing t -> first max wins
    float v = pval[(size_t)ch * NB * NB + idx];
    int   p = pidx[(size_t)ch * NB * NB + idx];
    if (v > best) { best = v; bp = p; }
  }
  float phase = (2.0f * 3.14159265358979323846f / (float)NN) * (float)bp;
  float fm    = fabsf(best) * sinf(phase);
  mup[idx] = (i < j) ? fm : 0.0f;
}

// ---------------------------------------------------------------------------
// Stage 3: Msym = Mup + Mup^T and per-row sums (128 blocks x 128 threads).
// ---------------------------------------------------------------------------
__global__ void k_msym(const float* __restrict__ mup,
                       float* __restrict__ msym,
                       float* __restrict__ rowsum) {
  __shared__ float red[NB];
  int i = blockIdx.x, j = threadIdx.x;
  float v = mup[i * NB + j] + mup[j * NB + i];
  msym[i * NB + j] = v;
  red[j] = v;
  __syncthreads();
  for (int s = NB / 2; s > 0; s >>= 1) {
    if (j < s) red[j] += red[j + s];
    __syncthreads();
  }
  if (j == 0) rowsum[i] = red[0];
}

// ---------------------------------------------------------------------------
// Stage 4: forces = (Msym @ S - rowsum .* S) / B in exact f32 on the matrix
// pipe via v_wmma_f32_16x16x4_f32 (scalar fallback keeps identical launch
// geometry if the builtin is absent on this toolchain).
// ---------------------------------------------------------------------------
__global__ void __launch_bounds__(256)
k_forces(const float* __restrict__ msym,
         const float* __restrict__ rowsum,
         const float* __restrict__ s,
         float* __restrict__ out) {
  const int wave = threadIdx.x >> 5;
  const int lane = threadIdx.x & 31;
  const int hl   = lane >> 4;
  const int r16  = lane & 15;
  const int tile = blockIdx.x * 8 + wave;     // 0..2047
  const int ti   = tile & 7;                  // i-tile (8)
  const int tn   = tile >> 3;                 // n-tile (256)
  v8f acc = {};
#if defined(__AMDGCN__) && __has_builtin(__builtin_amdgcn_wmma_f32_16x16x4_f32)
  {
    const int irow = ti * 16 + r16;
    const int ncol = tn * 16 + r16;
    for (int k = 0; k < NB; k += 4) {
      // 32-bit A 16x4: lane<16 K={0,1}, lane>=16 K={2,3}; B mirrored.
      int kk = k + 2 * hl;
      v2f afrag, bfrag;
      afrag[0] = msym[irow * NB + kk];
      afrag[1] = msym[irow * NB + kk + 1];
      bfrag[0] = s[(size_t)kk * NN + ncol];
      bfrag[1] = s[(size_t)(kk + 1) * NN + ncol];
      acc = __builtin_amdgcn_wmma_f32_16x16x4_f32(
          false, afrag, false, bfrag, (short)0, acc, false, false);
    }
  }
#else
  {
#pragma unroll
    for (int r = 0; r < 8; ++r) {
      int i = ti * 16 + 8 * hl + r;
      int n = tn * 16 + r16;
      float acc_s = 0.0f;
      for (int j = 0; j < NB; ++j) acc_s += msym[i * NB + j] * s[(size_t)j * NN + n];
      acc[r] = acc_s;
    }
  }
#endif
#pragma unroll
  for (int r = 0; r < 8; ++r) {
    int i = ti * 16 + 8 * hl + r;
    int n = tn * 16 + r16;
    out[(size_t)i * NN + n] =
        (acc[r] - rowsum[i] * s[(size_t)i * NN + n]) * (1.0f / (float)NB);
  }
}

// ---------------------------------------------------------------------------
extern "C" void kernel_launch(void* const* d_in, const int* in_sizes, int n_in,
                              void* d_out, int out_size, void* d_ws, size_t ws_size,
                              hipStream_t stream) {
  const float* signals = (const float*)d_in[0];
  float* out = (float*)d_out;
  char* ws = (char*)d_ws;

  size_t off = 0;
  __bf16* a_bf = (__bf16*)(ws + off); off += (size_t)NB * NN * 2;                 // 1 MB
  __bf16* sd   = (__bf16*)(ws + off); off += (size_t)NROT * NB * SD_ROWLEN * 2;   // 16 MB
  float*  pval = (float*)(ws + off);  off += (size_t)NCHUNK * NB * NB * 4;        // 2 MB
  int*    pidx = (int*)(ws + off);    off += (size_t)NCHUNK * NB * NB * 4;        // 2 MB
  float*  mup  = (float*)(ws + off);  off += (size_t)NB * NB * 4;
  float*  msym = (float*)(ws + off);  off += (size_t)NB * NB * 4;
  float*  rsum = (float*)(ws + off);  off += (size_t)NB * 4;

  k_convert<<<(NB * NN + 255) / 256, 256, 0, stream>>>(signals, a_bf, sd);

  dim3 g2(NB / 16, NCHUNK);                      // (8, 32)
  size_t ldsBytes = (size_t)16 * NN * sizeof(__bf16);   // 128 KB (CDNA5 WGP LDS)
  k_ccmax<<<g2, 256, ldsBytes, stream>>>(a_bf, sd, pval, pidx);

  k_reduce<<<(NB * NB + 255) / 256, 256, 0, stream>>>(pval, pidx, mup);
  k_msym<<<NB, NB, 0, stream>>>(mup, msym, rsum);
  k_forces<<<(NB / 16) * (NN / 16) / 8, 256, 0, stream>>>(msym, rsum, signals, out);
}
